// RNN_79577154060490
// MI455X (gfx1250) — matrix operand order repaired and checked by reading, compiled, not run on hardware
//
#include <hip/hip_runtime.h>

// Problem sizes (fixed by the reference)
#define B_  128
#define T_  100
#define NG_ 4096
#define NP_ 512

typedef __bf16 bf16;
typedef __attribute__((ext_vector_type(16))) __bf16 v16bf;
typedef __attribute__((ext_vector_type(8)))  __bf16 bf16x8;
typedef __attribute__((ext_vector_type(8)))  float  v8f;

union ABfrag { v16bf v; bf16x8 h[2]; };

#define KT         64                       // k-tile width (two WMMA k-slabs)
#define LDS_STRIDE 72                       // bf16 elems per row (144 B, 16B-aligned, bank-spread)
#define LDS_TILE   (64 * LDS_STRIDE)        // one 64x64 A tile (padded), elems
#define LDS_TILE_B (LDS_TILE * 2)           // bytes per tile buffer

// CDNA5 async memory->LDS copy (ASYNCcnt tracked), 16 bytes per lane.
// INST_OFFSET is applied to BOTH the LDS and the global address, so one
// address pair + one LDS-offset VGPR serves a whole 64-B stage.
__device__ __forceinline__ void async_ld_lds_b128(unsigned lds_byte_off,
                                                  const void* gaddr, int imm) {
    asm volatile("global_load_async_to_lds_b128 %0, %1, off offset:%2"
                 :: "v"(lds_byte_off), "v"(gaddr), "i"(imm));
}
__device__ __forceinline__ void wait_async0() {
    asm volatile("s_wait_asynccnt 0x0" ::: "memory");
}

// ---------------------------------------------------------------------------
// f32 -> bf16 conversion (one-time weight/activation prep)
// ---------------------------------------------------------------------------
__global__ void cvt_f32_bf16(const float* __restrict__ src,
                             bf16* __restrict__ dst, int n) {
    int i = blockIdx.x * blockDim.x + threadIdx.x;
    if (i < n) dst[i] = (bf16)src[i];
}

// ---------------------------------------------------------------------------
// Fused WMMA GEMM:  C[M,N] = A[M,K] @ Bw[N,K]^T  (+ mode-specific epilogue)
//   MODE 0: encoder  -> store bf16 C
//   MODE 1: RNN step -> C += v[:,t,:]@W_ih^T, tanh, store bf16
//   MODE 2: decoder  -> C += b_dec, store f32 to out[b][t][p] (row m = t*B+b)
//
// Grid:  (N/64, M/64).  Block: 128 threads = 4 wave32s.
// Each wave: one 16-wide N slice, 4 M-tiles of 16 -> 4 x v8f accumulators.
// Pipeline (64-wide k-tiles, double-buffered LDS, unrolled x2): async-stage
// A(k+1) + prefetch both B slabs into fresh regs while 8 WMMAs consume tile k;
// ONE s_wait_asynccnt + ONE barrier per 8 WMMAs / 16 ds_loads.
// ---------------------------------------------------------------------------
template<int MODE>
__global__ __launch_bounds__(128)
void gemm_bf16(const bf16* __restrict__ A,   // [M,K] row-major (bf16)
               const bf16* __restrict__ Bw,  // [N,K] row-major (bf16 weights)
               int K,
               bf16*  __restrict__ obf,      // MODE 0/1 output (bf16 [M,NG])
               float* __restrict__ of,       // MODE 2 output (f32 [B,T,NP])
               const float* __restrict__ v,    // MODE 1: [B,T,2]
               const float* __restrict__ Wih,  // MODE 1: [NG,2]
               int t,                          // MODE 1: timestep
               const float* __restrict__ bdec) // MODE 2: [NP]
{
    __shared__ __attribute__((aligned(16))) bf16 sA[2 * LDS_TILE]; // double buffer

    const int tid   = threadIdx.x;
    const int wave  = tid >> 5;
    const int lane  = tid & 31;
    const int lhalf = lane >> 4;   // 0: lanes 0-15, 1: lanes 16-31
    const int lmod  = lane & 15;

    const int n0 = blockIdx.x * 64 + wave * 16;  // this wave's N base
    const int m0 = blockIdx.y * 64;              // this block's M base

    v8f acc[4] = {};

    // B fragment source: lane's column n = n0 + lmod is weight row (n0+lmod)
    const bf16* brow = Bw + (size_t)(n0 + lmod) * K;

    // A staging: thread t copies 64 B of row (t>>1): elems (t&1)*32 .. +31
    const int      arow  = tid >> 1, ahalf = tid & 1;
    const bf16*    aptr  = A + (size_t)(m0 + arow) * K + ahalf * 32;
    const unsigned lbase = (unsigned)((arow * LDS_STRIDE + ahalf * 32) * sizeof(bf16));

    // stage one 64x64 A tile (k0..k0+63) into LDS buffer `buf` (4 async b128,
    // single address pair + single LDS-offset VGPR, immediate chunk offsets)
    auto stageA = [&](int k0, int buf) {
        unsigned l = lbase + (unsigned)buf * LDS_TILE_B;
        const bf16* g = aptr + k0;
        async_ld_lds_b128(l, g, 0);
        async_ld_lds_b128(l, g, 16);
        async_ld_lds_b128(l, g, 32);
        async_ld_lds_b128(l, g, 48);
    };
    // B fragment (32 k): lanes 0-15 hold K=0..7,16..23, lanes 16-31 hold K=8..15,24..31
    auto loadB = [&](int k0) {
        ABfrag r;
        const bf16* bb = brow + k0 + lhalf * 8;
        r.h[0] = *(const bf16x8*)(bb);
        r.h[1] = *(const bf16x8*)(bb + 16);
        return r;
    };
    // One 64-wide k-tile: 2 sub-slabs x (4 A frags into distinct regs + 4 WMMA)
    auto compute = [&](const ABfrag& bf0, const ABfrag& bf1, int buf) {
        const bf16* base = sA + buf * LDS_TILE;
        #pragma unroll
        for (int sub = 0; sub < 2; ++sub) {
            const ABfrag& bfr = sub ? bf1 : bf0;
            ABfrag af[4];
            #pragma unroll
            for (int mt = 0; mt < 4; ++mt) {
                const bf16* ar = base + (mt * 16 + lmod) * LDS_STRIDE + sub * 32;
                af[mt].h[0] = *(const bf16x8*)(ar + lhalf * 8);
                af[mt].h[1] = *(const bf16x8*)(ar + 16 + lhalf * 8);
            }
            #pragma unroll
            for (int mt = 0; mt < 4; ++mt)
                acc[mt] = __builtin_amdgcn_wmma_f32_16x16x32_bf16(
                    false, af[mt].v, false, bfr.v, (short)0, acc[mt], false, false);
        }
    };

    // ---- software pipeline over K (64-wide tiles; K/64 even: K in {512,4096})
    stageA(0, 0);
    ABfrag b0a = loadB(0), b0b = loadB(32);
    wait_async0();
    __syncthreads();

    int k0 = 0;
    for (; k0 + 2 * KT < K; k0 += 2 * KT) {
        stageA(k0 + KT, 1);                              // async copy overlaps compute
        ABfrag n0 = loadB(k0 + KT), n1 = loadB(k0 + KT + 32);
        compute(b0a, b0b, 0);
        wait_async0();
        __syncthreads();

        stageA(k0 + 2 * KT, 0);
        b0a = loadB(k0 + 2 * KT); b0b = loadB(k0 + 2 * KT + 32);
        compute(n0, n1, 1);
        wait_async0();
        __syncthreads();
    }
    {   // tail: two remaining 64-wide tiles (k0, k0+KT)
        stageA(k0 + KT, 1);
        ABfrag n0 = loadB(k0 + KT), n1 = loadB(k0 + KT + 32);
        compute(b0a, b0b, 0);
        wait_async0();
        __syncthreads();
        compute(n0, n1, 1);
    }

    // ------------------------- epilogue -------------------------
    const int n = n0 + lmod;   // this lane's output column

    if (MODE == 0) {
        #pragma unroll
        for (int mt = 0; mt < 4; ++mt)
            #pragma unroll
            for (int r = 0; r < 8; ++r) {
                int m = m0 + mt * 16 + r + lhalf * 8;
                obf[(size_t)m * NG_ + n] = (bf16)acc[mt][r];
            }
    } else if (MODE == 1) {
        const float wih0 = Wih[n * 2 + 0];
        const float wih1 = Wih[n * 2 + 1];
        #pragma unroll
        for (int mt = 0; mt < 4; ++mt)
            #pragma unroll
            for (int r = 0; r < 8; ++r) {
                int m = m0 + mt * 16 + r + lhalf * 8;   // batch index b
                float v0 = v[((size_t)m * T_ + t) * 2 + 0];
                float v1 = v[((size_t)m * T_ + t) * 2 + 1];
                float x  = acc[mt][r] + v0 * wih0 + v1 * wih1;
                obf[(size_t)m * NG_ + n] = (bf16)tanhf(x);
            }
    } else {
        const float bd = bdec[n];
        #pragma unroll
        for (int mt = 0; mt < 4; ++mt)
            #pragma unroll
            for (int r = 0; r < 8; ++r) {
                int m  = m0 + mt * 16 + r + lhalf * 8;  // m = t*B + b
                int tt = m >> 7;                        // / B_ (128)
                int b  = m & (B_ - 1);
                of[((size_t)b * T_ + tt) * NP_ + n] = acc[mt][r] + bd;
            }
    }
}

// ---------------------------------------------------------------------------
// Launch: convert weights once -> encoder GEMM -> 100 step GEMMs -> decoder
// ---------------------------------------------------------------------------
extern "C" void kernel_launch(void* const* d_in, const int* in_sizes, int n_in,
                              void* d_out, int out_size, void* d_ws, size_t ws_size,
                              hipStream_t stream) {
    const float* v    = (const float*)d_in[0];  // [B,T,2]
    const float* p0   = (const float*)d_in[1];  // [B,NP]
    const float* Wenc = (const float*)d_in[2];  // [NG,NP]
    const float* Wih  = (const float*)d_in[3];  // [NG,2]
    const float* Whh  = (const float*)d_in[4];  // [NG,NG]
    const float* Wdec = (const float*)d_in[5];  // [NP,NG]
    const float* bdec = (const float*)d_in[6];  // [NP]
    float* out = (float*)d_out;                 // [B,T,NP]

    char* ws = (char*)d_ws;
    bf16* Whh_b  = (bf16*)ws;  ws += (size_t)NG_ * NG_ * sizeof(bf16);
    bf16* Wenc_b = (bf16*)ws;  ws += (size_t)NG_ * NP_ * sizeof(bf16);
    bf16* Wdec_b = (bf16*)ws;  ws += (size_t)NP_ * NG_ * sizeof(bf16);
    bf16* p0_b   = (bf16*)ws;  ws += (size_t)B_  * NP_ * sizeof(bf16);
    bf16* h0_b   = (bf16*)ws;  ws += (size_t)B_  * NG_ * sizeof(bf16);
    bf16* gbuf   = (bf16*)ws;  // [T,B,NG] bf16: h at step t == decoder input

    auto cvt = [&](const float* s, bf16* d, int n) {
        cvt_f32_bf16<<<(n + 255) / 256, 256, 0, stream>>>(s, d, n);
    };
    cvt(Whh,  Whh_b,  NG_ * NG_);
    cvt(Wenc, Wenc_b, NG_ * NP_);
    cvt(Wdec, Wdec_b, NP_ * NG_);
    cvt(p0,   p0_b,   B_  * NP_);

    // Encoder: h0 = p0 @ W_enc^T   (M=128, N=4096, K=512)
    gemm_bf16<0><<<dim3(NG_ / 64, B_ / 64), 128, 0, stream>>>(
        p0_b, Wenc_b, NP_, h0_b, nullptr, nullptr, nullptr, 0, nullptr);

    // Recurrence: h_t = tanh(v_t @ W_ih^T + h_{t-1} @ W_hh^T)
    for (int t = 0; t < T_; ++t) {
        const bf16* hin  = (t == 0) ? h0_b : (gbuf + (size_t)(t - 1) * B_ * NG_);
        bf16*       hout = gbuf + (size_t)t * B_ * NG_;
        gemm_bf16<1><<<dim3(NG_ / 64, B_ / 64), 128, 0, stream>>>(
            hin, Whh_b, NG_, hout, nullptr, v, Wih, t, nullptr);
    }

    // Decoder: preds = g @ W_dec^T + b_dec   (M=T*B=12800, N=512, K=4096)
    gemm_bf16<2><<<dim3(NP_ / 64, (T_ * B_) / 64), 128, 0, stream>>>(
        gbuf, Wdec_b, NG_, nullptr, out, nullptr, nullptr, 0, bdec);
}
